// Decoder_68650757260123
// MI455X (gfx1250) — compile-verified, matrix operand order checked by reading, not run on hardware
//
#include <hip/hip_runtime.h>
#include <math.h>

// ---------------- problem constants ----------------
#define BB 2
#define SS 1024
#define DD 2048
#define HH 16
#define DHC 128
#define EE 8
#define HID 4096
#define TT (BB*SS)          // 2048 tokens
#define NBSD ((size_t)BB*SS*DD)

typedef __attribute__((ext_vector_type(16))) _Float16 v16h;
typedef __attribute__((ext_vector_type(8)))  float    v8f;

union H16 { v16h v; _Float16 h[16]; };
union F8  { v8f  v; float    f[8];  };

// ---- fragment gather: 16x32 f16 A/B tile from row-major [16][ld] f16 LDS ----
__device__ inline v16h ldfrag_lds(const _Float16* __restrict__ t, int ld, int lane) {
    int m = lane & 15, ks = (lane >> 4) & 1;
    const _Float16* r = t + m * ld + ks * 8;
    H16 u;
#pragma unroll
    for (int j = 0; j < 8; ++j) u.h[j] = r[j];
#pragma unroll
    for (int j = 0; j < 8; ++j) u.h[8 + j] = r[16 + j];
    return u.v;
}

// same layout, gathering from row-major f32 global memory (converts to f16)
__device__ inline v16h ldfrag_gf32(const float* __restrict__ row0, int ld, int kbase, int lane) {
    int m = lane & 15, ks = (lane >> 4) & 1;
    const float* r = row0 + (size_t)m * ld + kbase + ks * 8;
    H16 u;
#pragma unroll
    for (int j = 0; j < 8; ++j) u.h[j] = (_Float16)r[j];
#pragma unroll
    for (int j = 0; j < 8; ++j) u.h[8 + j] = (_Float16)r[16 + j];
    return u.v;
}

// ---------------- f32 -> f16 convert (4 elems/thread) ----------------
__global__ __launch_bounds__(256) void cvt_k(const float* __restrict__ a,
                                             _Float16* __restrict__ o, size_t n4) {
    size_t i = (size_t)blockIdx.x * 256 + threadIdx.x;
    if (i >= n4) return;
    float4 v = ((const float4*)a)[i];
    union { _Float16 h[4]; unsigned long long u; } p;
    p.h[0] = (_Float16)v.x; p.h[1] = (_Float16)v.y;
    p.h[2] = (_Float16)v.z; p.h[3] = (_Float16)v.w;
    ((unsigned long long*)o)[i] = p.u;
}

// ---------------- RMSNorm (f32 in, f16 out) ----------------
__global__ __launch_bounds__(256) void rmsnorm_k(const float* __restrict__ x,
                                                 const float* __restrict__ w,
                                                 _Float16* __restrict__ o, int Dd) {
    __shared__ float red[256];
    int r = blockIdx.x;
    const float* xr = x + (size_t)r * Dd;
    float s = 0.f;
    for (int i = threadIdx.x; i < Dd; i += 256) { float v = xr[i]; s += v * v; }
    red[threadIdx.x] = s; __syncthreads();
    for (int off = 128; off > 0; off >>= 1) {
        if (threadIdx.x < off) red[threadIdx.x] += red[threadIdx.x + off];
        __syncthreads();
    }
    float inv = rsqrtf(red[0] / (float)Dd + 1e-8f);
    for (int i = threadIdx.x; i < Dd; i += 256)
        o[(size_t)r * Dd + i] = (_Float16)(xr[i] * inv * w[i]);
}

// ---------------- WMMA GEMM, f16 operands, async global->LDS double buffer ----
// C[M,N] = A[M,K] * W[N,K]^T
// MODE 0: C = acc (+bias)   MODE 1: C = res + acc (+bias)
// MODE 2: C += rowscale[m*rs_stride] * acc
#define GBM 128
#define GBN 128
#define GBK 32
#define GLD 40   // padded f16 row stride (80B -> 16B-aligned rows)

// async-copy one 128x32 f16 tile (row-major, leading dim ldg) into LDS
__device__ inline void async_tile(const _Float16* __restrict__ g, int ldg,
                                  _Float16* ldst, int tid) {
    // 4 chunks of 16B per row, 512 chunks, 256 threads -> 2 each
#pragma unroll
    for (int it = 0; it < 2; ++it) {
        int i = tid + it * 256;
        int row = i >> 2, col = (i & 3) << 3;
        unsigned dst = (unsigned)(size_t)(ldst + row * GLD + col);
        unsigned long long src = (unsigned long long)(size_t)(g + (size_t)row * ldg + col);
        asm volatile("global_load_async_to_lds_b128 %0, %1, off"
                     :: "v"(dst), "v"(src) : "memory");
    }
}

template<int MODE, bool HASBIAS>
__global__ __launch_bounds__(256) void gemm_f16(const _Float16* __restrict__ A,
                                                const _Float16* __restrict__ W,
                                                const float* __restrict__ bias,
                                                const float* __restrict__ res,
                                                const float* __restrict__ rowscale,
                                                int rs_stride,
                                                float* __restrict__ C,
                                                int M, int N, int K) {
    __shared__ _Float16 As[2][GBM * GLD];
    __shared__ _Float16 Ws[2][GBN * GLD];

    const int tid  = threadIdx.x;
    const int lane = tid & 31;
    const int wid  = tid >> 5;            // 0..7
    const int wm   = wid >> 2;            // 0..1  (64-row slab)
    const int wn   = wid & 3;             // 0..3  (32-col slab)
    const int bm0  = blockIdx.y * GBM;
    const int bn0  = blockIdx.x * GBN;
    const _Float16* Ab = A + (size_t)bm0 * K;
    const _Float16* Wb = W + (size_t)bn0 * K;

    F8 acc[4][2];
#pragma unroll
    for (int mi = 0; mi < 4; ++mi)
#pragma unroll
        for (int ni = 0; ni < 2; ++ni)
#pragma unroll
            for (int j = 0; j < 8; ++j) acc[mi][ni].f[j] = 0.f;

    // prologue: prefetch k0 = 0 into buffer 0
    async_tile(Ab, K, As[0], tid);
    async_tile(Wb, K, Ws[0], tid);
    asm volatile("s_wait_asynccnt 0" ::: "memory");
    __syncthreads();

    int buf = 0;
    for (int k0 = 0; k0 < K; k0 += GBK) {
        int nbuf = buf ^ 1;
        if (k0 + GBK < K) {   // prefetch next tile while computing this one
            async_tile(Ab + (k0 + GBK), K, As[nbuf], tid);
            async_tile(Wb + (k0 + GBK), K, Ws[nbuf], tid);
        }

        v16h bfr[2];
#pragma unroll
        for (int ni = 0; ni < 2; ++ni)
            bfr[ni] = ldfrag_lds(Ws[buf] + (wn * 32 + ni * 16) * GLD, GLD, lane);
#pragma unroll
        for (int mi = 0; mi < 4; ++mi) {
            v16h afr = ldfrag_lds(As[buf] + (wm * 64 + mi * 16) * GLD, GLD, lane);
#pragma unroll
            for (int ni = 0; ni < 2; ++ni)
                acc[mi][ni].v = __builtin_amdgcn_wmma_f32_16x16x32_f16(
                    false, afr, false, bfr[ni], (short)0, acc[mi][ni].v, false, false);
        }

        asm volatile("s_wait_asynccnt 0" ::: "memory");
        __syncthreads();
        buf = nbuf;
    }

    const int lanelo = lane & 15, hf = lane >> 4;
#pragma unroll
    for (int mi = 0; mi < 4; ++mi)
#pragma unroll
        for (int ni = 0; ni < 2; ++ni)
#pragma unroll
            for (int j = 0; j < 8; ++j) {
                int m = bm0 + wm * 64 + mi * 16 + j + hf * 8;
                int n = bn0 + wn * 32 + ni * 16 + lanelo;
                float val = acc[mi][ni].f[j];
                size_t ci = (size_t)m * N + n;
                if (MODE == 2) {
                    C[ci] += rowscale[(size_t)m * rs_stride] * val;
                } else {
                    if (HASBIAS) val += bias[n];
                    if (MODE == 1) val += res[ci];
                    C[ci] = val;
                }
            }
}

// ---------------- RoPE (reference quirk: angle = head_index * inv_freq) --------
__global__ __launch_bounds__(256) void rope_k(float* __restrict__ q, int total) {
    int i = blockIdx.x * 256 + threadIdx.x;
    if (i >= total) return;
    int dh2 = i & 63;               // DH/2 = 64
    int h   = (i >> 6) & 15;
    int s   = (i >> 10) & (SS - 1);
    int b   = i >> 20;
    float inv = __expf(-((float)(2 * dh2) / (float)DHC) * 9.210340371976184f); // ln(10000)
    float ang = (float)h * inv;
    float c = cosf(ang), sn = sinf(ang);
    size_t base = ((size_t)(b * SS + s)) * DD + h * DHC + 2 * dh2;
    float x1 = q[base], x2 = q[base + 1];
    q[base]     = x1 * c - x2 * sn;
    q[base + 1] = x1 * sn + x2 * c;
}

// ---------------- Flash attention (causal) with WMMA; f16 output -------------
#define LDK 136
#define LDV 72
#define LDP 72

__global__ __launch_bounds__(128) void flash_attn_k(const float* __restrict__ qg,
                                                    const float* __restrict__ kg,
                                                    const float* __restrict__ vg,
                                                    _Float16* __restrict__ og) {
    __shared__ _Float16 Ks[64 * LDK];
    __shared__ _Float16 Vt[DHC * LDV];
    __shared__ _Float16 Pl[4 * 16 * LDP];

    const int tid  = threadIdx.x;
    const int lane = tid & 31;
    const int wid  = tid >> 5;            // 0..3
    const int lanelo = lane & 15, hf = lane >> 4;
    const int b = blockIdx.y / HH;
    const int h = blockIdx.y % HH;
    const int q0 = blockIdx.x * 64;
    const int m0 = q0 + wid * 16;

    const float* qrow0 = qg + ((size_t)(b * SS + m0)) * DD + h * DHC;
    v16h qf[4];
#pragma unroll
    for (int f = 0; f < 4; ++f) qf[f] = ldfrag_gf32(qrow0, DD, f * 32, lane);

    F8 oacc[8];
    float mx[8], l[8];
#pragma unroll
    for (int t = 0; t < 8; ++t) {
#pragma unroll
        for (int j = 0; j < 8; ++j) oacc[t].f[j] = 0.f;
        mx[t] = -INFINITY; l[t] = 0.f;
    }

    const float sscale = 0.08838834764831845f;   // 1/sqrt(128)

    for (int kb0 = 0; kb0 <= q0; kb0 += 64) {
        __syncthreads();
        for (int i = tid; i < (64 * DHC) / 4; i += 128) {
            int row = i >> 5;
            int col = (i & 31) << 2;
            float4 k4 = *(const float4*)(kg + ((size_t)(b * SS + kb0 + row)) * DD + h * DHC + col);
            _Float16* d = Ks + row * LDK + col;
            d[0] = (_Float16)k4.x; d[1] = (_Float16)k4.y;
            d[2] = (_Float16)k4.z; d[3] = (_Float16)k4.w;
            float4 v4 = *(const float4*)(vg + ((size_t)(b * SS + kb0 + row)) * DD + h * DHC + col);
            Vt[(col + 0) * LDV + row] = (_Float16)v4.x;
            Vt[(col + 1) * LDV + row] = (_Float16)v4.y;
            Vt[(col + 2) * LDV + row] = (_Float16)v4.z;
            Vt[(col + 3) * LDV + row] = (_Float16)v4.w;
        }
        __syncthreads();

        F8 sc[4];
#pragma unroll
        for (int nt = 0; nt < 4; ++nt)
#pragma unroll
            for (int j = 0; j < 8; ++j) sc[nt].f[j] = 0.f;
#pragma unroll
        for (int f = 0; f < 4; ++f) {
#pragma unroll
            for (int nt = 0; nt < 4; ++nt) {
                v16h bfr = ldfrag_lds(Ks + (nt * 16) * LDK + f * 32, LDK, lane);
                sc[nt].v = __builtin_amdgcn_wmma_f32_16x16x32_f16(
                    false, qf[f], false, bfr, (short)0, sc[nt].v, false, false);
            }
        }

#pragma unroll
        for (int nt = 0; nt < 4; ++nt)
#pragma unroll
            for (int j = 0; j < 8; ++j) {
                float v = sc[nt].f[j] * sscale;
                int key = kb0 + nt * 16 + lanelo;
                int qr  = m0 + j + hf * 8;
                sc[nt].f[j] = (key > qr) ? -INFINITY : v;
            }

#pragma unroll
        for (int j = 0; j < 8; ++j) {
            float cur = sc[0].f[j];
#pragma unroll
            for (int nt = 1; nt < 4; ++nt) cur = fmaxf(cur, sc[nt].f[j]);
            for (int d = 8; d >= 1; d >>= 1) cur = fmaxf(cur, __shfl_xor(cur, d, 16));
            float nm  = fmaxf(mx[j], cur);
            float scl = (mx[j] == -INFINITY) ? 0.f : __expf(mx[j] - nm);
            mx[j] = nm;
            float ps = 0.f;
#pragma unroll
            for (int nt = 0; nt < 4; ++nt) {
                float p = __expf(sc[nt].f[j] - nm);
                sc[nt].f[j] = p;
                ps += p;
            }
            for (int d = 8; d >= 1; d >>= 1) ps += __shfl_xor(ps, d, 16);
            l[j] = l[j] * scl + ps;
#pragma unroll
            for (int t = 0; t < 8; ++t) oacc[t].f[j] *= scl;
        }

        _Float16* Pw = Pl + wid * 16 * LDP;
#pragma unroll
        for (int nt = 0; nt < 4; ++nt)
#pragma unroll
            for (int j = 0; j < 8; ++j)
                Pw[(j + hf * 8) * LDP + nt * 16 + lanelo] = (_Float16)sc[nt].f[j];
        asm volatile("s_wait_dscnt 0" ::: "memory");

        v16h pa[2];
#pragma unroll
        for (int c = 0; c < 2; ++c) pa[c] = ldfrag_lds(Pw + c * 32, LDP, lane);
#pragma unroll
        for (int t = 0; t < 8; ++t)
#pragma unroll
            for (int c = 0; c < 2; ++c) {
                v16h bfr = ldfrag_lds(Vt + (t * 16) * LDV + c * 32, LDV, lane);
                oacc[t].v = __builtin_amdgcn_wmma_f32_16x16x32_f16(
                    false, pa[c], false, bfr, (short)0, oacc[t].v, false, false);
            }
    }

#pragma unroll
    for (int t = 0; t < 8; ++t)
#pragma unroll
        for (int j = 0; j < 8; ++j) {
            int m = m0 + j + hf * 8;
            og[((size_t)(b * SS + m)) * DD + h * DHC + t * 16 + lanelo] =
                (_Float16)(oacc[t].f[j] / l[j]);
        }
}

// ---------------- router: logits[t,e] = xn2[t] . w_router[e] + b ------------
__global__ __launch_bounds__(256) void router_k(const _Float16* __restrict__ xn,
                                                const float* __restrict__ wr,
                                                const float* __restrict__ br,
                                                float* __restrict__ logits) {
    __shared__ float sm[EE * 256];
    int t = blockIdx.x;
    float acc[EE];
#pragma unroll
    for (int e = 0; e < EE; ++e) acc[e] = 0.f;
    const _Float16* xr = xn + (size_t)t * DD;
    for (int i = threadIdx.x; i < DD; i += 256) {
        float xv = (float)xr[i];
#pragma unroll
        for (int e = 0; e < EE; ++e) acc[e] += xv * wr[(size_t)e * DD + i];
    }
#pragma unroll
    for (int e = 0; e < EE; ++e) sm[e * 256 + threadIdx.x] = acc[e];
    __syncthreads();
    for (int off = 128; off > 0; off >>= 1) {
        if (threadIdx.x < off)
#pragma unroll
            for (int e = 0; e < EE; ++e)
                sm[e * 256 + threadIdx.x] += sm[e * 256 + threadIdx.x + off];
        __syncthreads();
    }
    if (threadIdx.x < EE)
        logits[(size_t)t * EE + threadIdx.x] = sm[threadIdx.x * 256] + br[threadIdx.x];
}

// ---------------- top-2 softmax weights ----------------
__global__ __launch_bounds__(256) void topk_k(const float* __restrict__ logits,
                                              float* __restrict__ wts, int T) {
    int t = blockIdx.x * 256 + threadIdx.x;
    if (t >= T) return;
    float v[EE];
#pragma unroll
    for (int e = 0; e < EE; ++e) v[e] = logits[(size_t)t * EE + e];
    int i1 = 0;
#pragma unroll
    for (int e = 1; e < EE; ++e) if (v[e] > v[i1]) i1 = e;
    int i2 = (i1 == 0) ? 1 : 0;
#pragma unroll
    for (int e = 0; e < EE; ++e) if (e != i1 && v[e] > v[i2]) i2 = e;
    float m  = fmaxf(v[i1], v[i2]);
    float e1 = __expf(v[i1] - m), e2 = __expf(v[i2] - m);
    float d  = e1 + e2;
#pragma unroll
    for (int e = 0; e < EE; ++e)
        wts[(size_t)t * EE + e] = (e == i1) ? e1 / d : ((e == i2) ? e2 / d : 0.f);
}

// ---------------- elementwise ----------------
__global__ __launch_bounds__(256) void silu_mul_k(const float* __restrict__ g,
                                                  const float* __restrict__ u,
                                                  _Float16* __restrict__ a, size_t n) {
    size_t i = (size_t)blockIdx.x * 256 + threadIdx.x;
    if (i >= n) return;
    float gv = g[i];
    a[i] = (_Float16)(gv * (1.f / (1.f + __expf(-gv))) * u[i]);
}

__global__ __launch_bounds__(256) void copy_k(const float* __restrict__ a,
                                              float* __restrict__ o, size_t n) {
    size_t i = (size_t)blockIdx.x * 256 + threadIdx.x;
    if (i < n) o[i] = a[i];
}

// ---------------- launcher ----------------
extern "C" void kernel_launch(void* const* d_in, const int* in_sizes, int n_in,
                              void* d_out, int out_size, void* d_ws, size_t ws_size,
                              hipStream_t stream) {
    const float* x    = (const float*)d_in[0];
    const float* wq   = (const float*)d_in[1];
    const float* bq   = (const float*)d_in[2];
    const float* wk   = (const float*)d_in[3];
    const float* bk   = (const float*)d_in[4];
    const float* wv   = (const float*)d_in[5];
    const float* bv   = (const float*)d_in[6];
    const float* wo   = (const float*)d_in[7];
    const float* bo   = (const float*)d_in[8];
    const float* wrt  = (const float*)d_in[9];
    const float* brt  = (const float*)d_in[10];
    const float* wg   = (const float*)d_in[11];
    const float* wu   = (const float*)d_in[12];
    const float* wd   = (const float*)d_in[13];
    const float* rms1 = (const float*)d_in[14];
    const float* rms2 = (const float*)d_in[15];
    float* out = (float*)d_out;

    // f32 workspace region
    float* ws     = (float*)d_ws;
    float* qb     = ws + 0 * NBSD;
    float* kbuf   = ws + 1 * NBSD;
    float* vbuf   = ws + 2 * NBSD;
    float* xattn  = ws + 3 * NBSD;
    float* gbuf   = ws + 4 * NBSD;        // T*HID = 2*NBSD
    float* ubuf   = ws + 6 * NBSD;        // T*HID
    float* logits = ws + 8 * NBSD;        // T*E
    float* wts    = logits + (size_t)TT * EE;
    // f16 workspace region
    _Float16* hb   = (_Float16*)(ws + 9 * NBSD);
    _Float16* xnH  = hb + 0 * NBSD;
    _Float16* obH  = hb + 1 * NBSD;
    _Float16* xn2H = hb + 2 * NBSD;
    _Float16* wqH  = hb + 3 * NBSD;
    _Float16* wkH  = hb + 4 * NBSD;
    _Float16* wvH  = hb + 5 * NBSD;
    _Float16* woH  = hb + 6 * NBSD;
    _Float16* wgH  = hb + 7 * NBSD;       // HID*D = 2*NBSD halves
    _Float16* wuH  = hb + 9 * NBSD;
    _Float16* wdH  = hb + 11 * NBSD;
    _Float16* aH   = hb + 13 * NBSD;      // T*HID halves

    dim3 gblk(256);
    dim3 gemmDD(DD / GBN, TT / GBM);      // N=2048, M=2048
    dim3 gemmHID(HID / GBN, TT / GBM);    // N=4096
    size_t cvtDD  = NBSD / 4;             // float4 chunks for DxD / BSxD
    size_t cvtGU  = (2 * NBSD) / 4;       // HIDxD chunks
    int cvtDDg = (int)((cvtDD + 255) / 256), cvtGUg = (int)((cvtGU + 255) / 256);

    // 0) one-shot f16 conversion of the attention weight panels
    cvt_k<<<cvtDDg, gblk, 0, stream>>>(wq, wqH, cvtDD);
    cvt_k<<<cvtDDg, gblk, 0, stream>>>(wk, wkH, cvtDD);
    cvt_k<<<cvtDDg, gblk, 0, stream>>>(wv, wvH, cvtDD);
    cvt_k<<<cvtDDg, gblk, 0, stream>>>(wo, woH, cvtDD);
    // 1) rmsnorm -> f16 activations
    rmsnorm_k<<<TT, gblk, 0, stream>>>(x, rms1, xnH, DD);
    // 2) q,k,v projections (f16 operands, async double-buffered)
    gemm_f16<0, true><<<gemmDD, gblk, 0, stream>>>(xnH, wqH, bq, nullptr, nullptr, 0, qb,   TT, DD, DD);
    gemm_f16<0, true><<<gemmDD, gblk, 0, stream>>>(xnH, wkH, bk, nullptr, nullptr, 0, kbuf, TT, DD, DD);
    gemm_f16<0, true><<<gemmDD, gblk, 0, stream>>>(xnH, wvH, bv, nullptr, nullptr, 0, vbuf, TT, DD, DD);
    // 3) rope on q,k
    int npairs = BB * SS * HH * (DHC / 2);
    rope_k<<<(npairs + 255) / 256, gblk, 0, stream>>>(qb, npairs);
    rope_k<<<(npairs + 255) / 256, gblk, 0, stream>>>(kbuf, npairs);
    // 4) attention -> f16 O
    flash_attn_k<<<dim3(SS / 64, BB * HH), dim3(128), 0, stream>>>(qb, kbuf, vbuf, obH);
    // 5) output projection + residual
    gemm_f16<1, true><<<gemmDD, gblk, 0, stream>>>(obH, woH, bo, x, nullptr, 0, xattn, TT, DD, DD);
    // 6) rmsnorm2 -> f16
    rmsnorm_k<<<TT, gblk, 0, stream>>>(xattn, rms2, xn2H, DD);
    // 7) router + top-2
    router_k<<<TT, gblk, 0, stream>>>(xn2H, wrt, brt, logits);
    topk_k<<<(TT + 255) / 256, gblk, 0, stream>>>(logits, wts, TT);
    // 8) seed output with attention residual
    copy_k<<<(int)((NBSD + 255) / 256), gblk, 0, stream>>>(xattn, out, NBSD);
    // 9) MoE experts (dense, as in reference)
    size_t nGU = (size_t)TT * HID;
    for (int e = 0; e < EE; ++e) {
        cvt_k<<<cvtGUg, gblk, 0, stream>>>(wg + (size_t)e * HID * DD, wgH, cvtGU);
        cvt_k<<<cvtGUg, gblk, 0, stream>>>(wu + (size_t)e * HID * DD, wuH, cvtGU);
        cvt_k<<<cvtGUg, gblk, 0, stream>>>(wd + (size_t)e * DD * HID, wdH, cvtGU);
        gemm_f16<0, false><<<gemmHID, gblk, 0, stream>>>(xn2H, wgH, nullptr, nullptr, nullptr, 0, gbuf, TT, HID, DD);
        gemm_f16<0, false><<<gemmHID, gblk, 0, stream>>>(xn2H, wuH, nullptr, nullptr, nullptr, 0, ubuf, TT, HID, DD);
        silu_mul_k<<<(int)((nGU + 255) / 256), gblk, 0, stream>>>(gbuf, ubuf, aH, nGU);
        gemm_f16<2, false><<<gemmDD, gblk, 0, stream>>>(aH, wdH, nullptr, nullptr, wts + e, EE, out, TT, DD, HID);
    }
}
